// CALayer_75574244540697
// MI455X (gfx1250) — matrix operand order repaired and checked by reading, not compiled, not run on hardware
//
#include <hip/hip_runtime.h>
#include <math.h>

typedef __attribute__((ext_vector_type(16))) _Float16 v16h;
typedef __attribute__((ext_vector_type(8)))  _Float16 v8h;
typedef __attribute__((ext_vector_type(8)))  float    v8f;

union V16H { v16h v; v8h h[2]; _Float16 e[16]; };

// f16 WMMA has no A/B negation (NEG bits reserved); plain accumulate only.
#define WMMA_F32_F16(A, B, C) \
  __builtin_amdgcn_wmma_f32_16x16x32_f16(false, (A), false, (B), (short)0, (C), false, false)

// ---------------- workspace layout (bytes) ----------------
// gt  : 1024 img * 2 planes * 7 mtiles * 8 ksteps * 512 halfs (A-operand blocked fp16)
static constexpr size_t GT_HALFS  = 1024ull * 2 * 7 * 8 * 512;  // 58,720,256
static constexpr size_t GT_BYTES  = GT_HALFS * 2;               // 117,440,512
static constexpr size_t TW1_HALFS = 2ull * 7 * 8 * 512;         // 57,344  (cos, -sin)
static constexpr size_t TW2_HALFS = 3ull * 8 * 7 * 512;         // 86,016  (cos, -sin, +sin)
static constexpr size_t TW1_OFF   = GT_BYTES;
static constexpr size_t TW2_OFF   = TW1_OFF + TW1_HALFS * 2;
static constexpr size_t Y_OFF     = TW2_OFF + TW2_HALFS * 2;    // 1024 floats
static constexpr size_t S_OFF     = Y_OFF + 4096;               // 1024 floats

// Operand-slot index maps (this code's contiguous k order k = 16*(lane/16) + j, same for A and B; wave32)
__device__ __forceinline__ int kmapA(int lane, int j) {
  return ((lane >> 4) << 4) + j;                            // A 16x32 f16
}
__device__ __forceinline__ int kmapB(int lane, int j) {
  return ((lane >> 4) << 4) + j;                            // B 32x16 f16
}

// ---------------- kernel 0: twiddle tables in operand layout ----------------
__global__ void build_tw(_Float16* __restrict__ tw1, _Float16* __restrict__ tw2) {
  int idx = blockIdx.x * blockDim.x + threadIdx.x;          // [0, 143360)
  if (idx >= (int)(TW1_HALFS + TW2_HALFS)) return;
  const float c0 = -6.283185307179586f / 256.0f;            // -2*pi/256
  if (idx < (int)TW1_HALFS) {
    // tw1[(p*7+mt)*8+ks][lane][j] : A operand of row-DFT (rows u, cols h)
    int f = idx;
    int j = f & 15, l = (f >> 4) & 31, blk = f >> 9;        // blk in [0,112)
    int ks = blk & 7, pm = blk >> 3, mt = pm % 7, p = pm / 7;
    int r = mt * 16 + (l & 15);
    int k = ks * 32 + kmapA(l, j);
    float val = 0.0f;
    if (r < 100) {
      int fk = ((r - 50) * k) % 256;                        // exact range reduction
      float s, c; sincosf(c0 * (float)fk, &s, &c);
      val = (p == 0) ? c : s;                               // cos / -sin(2*pi*f*k/256)
    }
    tw1[f] = (_Float16)val;
  } else {
    // tw2[(p*8+ks)*7+nt][lane][j] : B operand of col-DFT (rows w, cols v)
    // p=0: cos, p=1: -sin (imag part), p=2: +sin (negated imag, for Fre)
    int f = idx - (int)TW1_HALFS;
    int j = f & 15, l = (f >> 4) & 31, blk = f >> 9;        // blk in [0,168)
    int nt = blk % 7, pk = blk / 7, ks = pk & 7, p = pk >> 3;
    int n = nt * 16 + (l & 15);
    int k = ks * 32 + kmapB(l, j);
    float val = 0.0f;
    if (n < 100) {
      int fk = ((n - 50) * k) % 256;
      float s, c; sincosf(c0 * (float)fk, &s, &c);
      val = (p == 0) ? c : ((p == 1) ? s : -s);
    }
    tw2[f] = (_Float16)val;
  }
}

// ---------------- kernel 1: row DFT  G = TW1 @ x  (per image, per 64-col chunk) ----
__global__ __launch_bounds__(224) void dft_rows(const float* __restrict__ x,
                                                const _Float16* __restrict__ tw1,
                                                _Float16* __restrict__ gt) {
  __shared__ __align__(16) _Float16 xs[64][264];    // x transposed: [w_local][h], 16B rows
  __shared__ _Float16 gout[2][64][112];             // [plane][w_local][m]

  const int img   = blockIdx.x;                     // 0..1023
  const int chunk = blockIdx.y;                     // 0..3 (w chunk of 64)
  const int tid   = threadIdx.x;
  const float* xi = x + (size_t)img * 65536 + chunk * 64;

  for (int i = tid; i < 64 * 256; i += 224) {       // coalesced global, 2-way LDS conflict
    int h = i >> 6, c = i & 63;
    xs[c][h] = (_Float16)xi[h * 256 + c];
  }
  __syncthreads();

  const int lane = tid & 31, mt = tid >> 5;         // 7 waves -> 7 m-tiles
  const int hl = lane >> 4, ln = lane & 15;

  v8f accRe[4], accIm[4];
  const v8f vzero = {0.f, 0.f, 0.f, 0.f, 0.f, 0.f, 0.f, 0.f};
  for (int nt = 0; nt < 4; ++nt) { accRe[nt] = vzero; accIm[nt] = vzero; }

  const _Float16* tc = tw1 + (size_t)(mt * 8) * 512;        // cos plane
  const _Float16* ts = tw1 + (size_t)((7 + mt) * 8) * 512;  // -sin plane
  for (int ks = 0; ks < 8; ++ks) {
    v16h aC = *(const v16h*)(tc + ks * 512 + lane * 16);
    v16h aS = *(const v16h*)(ts + ks * 512 + lane * 16);
    int kb = ks * 32 + hl * 16;                     // B slots j=0..15 contiguous in h
    for (int nt = 0; nt < 4; ++nt) {
      V16H b;
      int c = nt * 16 + ln;
      b.h[0] = *(const v8h*)&xs[c][kb];
      b.h[1] = *(const v8h*)&xs[c][kb + 8];
      accRe[nt] = WMMA_F32_F16(aC, b.v, accRe[nt]);
      accIm[nt] = WMMA_F32_F16(aS, b.v, accIm[nt]);
    }
  }
  __syncthreads();

  for (int nt = 0; nt < 4; ++nt) {
#pragma unroll
    for (int r = 0; r < 8; ++r) {                   // D layout: m = r + 8*hl
      int m = mt * 16 + r + hl * 8;
      int wl = nt * 16 + ln;
      gout[0][wl][m] = (_Float16)accRe[nt][r];
      gout[1][wl][m] = (_Float16)accIm[nt][r];
    }
  }
  __syncthreads();

  // Re-block G into WMMA A-operand layout for kernel 2; coalesced global stores.
  for (int f = tid; f < 14336; f += 224) {          // exactly 64 iters
    int j = f & 15, l = (f >> 4) & 31;
    int rest = f >> 9;                              // [0,28)
    int ksl = rest & 1, pm = rest >> 1;             // [0,14)
    int mtt = pm % 7, p = pm / 7;
    int m  = mtt * 16 + (l & 15);
    int wl = ksl * 32 + kmapA(l, j);
    int ksg = chunk * 2 + ksl;
    gt[((((size_t)img * 2 + p) * 7 + mtt) * 8 + ksg) * 512 + l * 16 + j] = gout[p][wl][m];
  }
}

// ---------------- kernel 2: col DFT (complex) + |F| + mean -----------------
__global__ __launch_bounds__(224) void dft_cols_mag(const _Float16* __restrict__ gt,
                                                    const _Float16* __restrict__ tw2,
                                                    float* __restrict__ y) {
  const int img = blockIdx.x;
  const int tid = threadIdx.x, lane = tid & 31, mt = tid >> 5;

  v8f fre[7], fim[7];
  const v8f vzero = {0.f, 0.f, 0.f, 0.f, 0.f, 0.f, 0.f, 0.f};
  for (int nt = 0; nt < 7; ++nt) { fre[nt] = vzero; fim[nt] = vzero; }

  const _Float16* Are = gt + (((size_t)img * 2 + 0) * 7 + mt) * 8 * 512;
  const _Float16* Aim = gt + (((size_t)img * 2 + 1) * 7 + mt) * 8 * 512;

  for (int ks = 0; ks < 8; ++ks) {
    v16h aRe = *(const v16h*)(Are + ks * 512 + lane * 16);  // contiguous 32B/lane
    v16h aIm = *(const v16h*)(Aim + ks * 512 + lane * 16);
    for (int nt = 0; nt < 7; ++nt) {
      v16h bC  = *(const v16h*)(tw2 + ((size_t)(( 0 + ks) * 7 + nt) * 32 + lane) * 16);
      v16h bS  = *(const v16h*)(tw2 + ((size_t)(( 8 + ks) * 7 + nt) * 32 + lane) * 16);
      v16h bNS = *(const v16h*)(tw2 + ((size_t)((16 + ks) * 7 + nt) * 32 + lane) * 16);
      // F = (Gre + i Gim)(C + i S):  Fre = Gre*C + Gim*(-S) ; Fim = Gre*S + Gim*C
      fre[nt] = WMMA_F32_F16(aRe, bC,  fre[nt]);
      fre[nt] = WMMA_F32_F16(aIm, bNS, fre[nt]);
      fim[nt] = WMMA_F32_F16(aRe, bS,  fim[nt]);
      fim[nt] = WMMA_F32_F16(aIm, bC,  fim[nt]);
    }
  }

  float sum = 0.0f;                                 // padded rows/cols are exactly 0
  for (int nt = 0; nt < 7; ++nt)
#pragma unroll
    for (int r = 0; r < 8; ++r)
      sum += sqrtf(fre[nt][r] * fre[nt][r] + fim[nt][r] * fim[nt][r]);

  for (int off = 16; off > 0; off >>= 1) sum += __shfl_down(sum, off);
  __shared__ float red[7];
  if (lane == 0) red[mt] = sum;
  __syncthreads();
  if (tid == 0) {
    float t = 0.f;
    for (int w = 0; w < 7; ++w) t += red[w];
    y[img] = t * (1.0f / 10000.0f);                 // mean over 100x100 crop
  }
}

// ---------------- kernel 3: squeeze-excite MLP (tiny) -----------------------
__global__ void mlp_kernel(const float* __restrict__ y,
                           const float* __restrict__ w1, const float* __restrict__ b1,
                           const float* __restrict__ w2, const float* __restrict__ b2,
                           float* __restrict__ s) {
  int t = blockIdx.x * blockDim.x + threadIdx.x;
  if (t >= 1024) return;
  int b = t >> 6, c = t & 63;
  float acc = b2[c];
  for (int m = 0; m < 4; ++m) {
    float hm = b1[m];
    for (int k = 0; k < 64; ++k) hm += w1[m * 64 + k] * y[b * 64 + k];
    hm = fmaxf(hm, 0.0f);
    acc += w2[c * 4 + m] * hm;
  }
  s[t] = 1.0f / (1.0f + expf(-acc));
}

// ---------------- kernel 4: broadcast scale (bandwidth pass) ----------------
__global__ void scale_kernel(const float4* __restrict__ x, const float* __restrict__ s,
                             float4* __restrict__ out) {
  size_t i = (size_t)blockIdx.x * blockDim.x + threadIdx.x;   // 16,777,216 float4
  float sc = s[i >> 14];                                       // 16384 float4 / image
  float4 v = x[i];
  v.x *= sc; v.y *= sc; v.z *= sc; v.w *= sc;
  out[i] = v;
}

extern "C" void kernel_launch(void* const* d_in, const int* in_sizes, int n_in,
                              void* d_out, int out_size, void* d_ws, size_t ws_size,
                              hipStream_t stream) {
  (void)in_sizes; (void)n_in; (void)out_size; (void)ws_size;
  const float* x  = (const float*)d_in[0];
  const float* w1 = (const float*)d_in[1];
  const float* b1 = (const float*)d_in[2];
  const float* w2 = (const float*)d_in[3];
  const float* b2 = (const float*)d_in[4];
  float* out = (float*)d_out;
  char* ws = (char*)d_ws;

  _Float16* gt  = (_Float16*)ws;
  _Float16* tw1 = (_Float16*)(ws + TW1_OFF);
  _Float16* tw2 = (_Float16*)(ws + TW2_OFF);
  float*    y   = (float*)(ws + Y_OFF);
  float*    s   = (float*)(ws + S_OFF);

  build_tw<<<560, 256, 0, stream>>>(tw1, tw2);                       // 143,360 elems
  dft_rows<<<dim3(1024, 4), 224, 0, stream>>>(x, tw1, gt);
  dft_cols_mag<<<1024, 224, 0, stream>>>(gt, tw2, y);
  mlp_kernel<<<4, 256, 0, stream>>>(y, w1, b1, w2, b2, s);
  scale_kernel<<<65536, 256, 0, stream>>>((const float4*)x, s, (float4*)out);
}
